// SemanticMapTokenizer_20521353740697
// MI455X (gfx1250) — compile-verified, matrix operand order checked by reading, not run on hardware
//
#include <hip/hip_runtime.h>

typedef __attribute__((ext_vector_type(16))) _Float16 v16h;
typedef __attribute__((ext_vector_type(8)))  _Float16 v8h;
typedef __attribute__((ext_vector_type(8)))  float    v8f;

#define NUM_CLASSES 256
#define K_DIM 256             // classes
#define N_DIM 1024            // embed dim
#define M_TOK 2048            // 2 * 32 * 32 tokens

// ---------------------------------------------------------------------------
// Kernel 1: convert embed_table f32 -> f16 and swizzle into WMMA B-fragment
// layout. B is K x N = 256 x 1024. Fragment (kt in [0,8), nt in [0,64)):
//   lanes 0-15  : n = lane,    halves 0..7 -> K=kb..kb+7,  8..15 -> K=kb+16..+23
//   lanes 16-31 : n = lane-16, halves 0..7 -> K=kb+8..+15, 8..15 -> K=kb+24..+31
// Stored per-lane contiguous: frag*512 + lane*16 halves -> one aligned v16h.
// ---------------------------------------------------------------------------
__global__ __launch_bounds__(256) void swizzle_b_kernel(
    const float* __restrict__ embed, _Float16* __restrict__ bswz) {
  int lane = threadIdx.x & 31;
  int frag = blockIdx.x * 8 + (threadIdx.x >> 5);   // 64 blocks * 8 waves = 512 frags
  int kt = frag >> 6;
  int nt = frag & 63;
  int n = nt * 16 + (lane & 15);
  int kbase = kt * 32 + ((lane & 16) ? 8 : 0);
  v16h out;
#pragma unroll
  for (int i = 0; i < 8; ++i) {
    out[i]     = (_Float16)embed[(kbase + i) * N_DIM + n];
    out[i + 8] = (_Float16)embed[(kbase + 16 + i) * N_DIM + n];
  }
  *(v16h*)(bswz + (size_t)frag * 512 + lane * 16) = out;
}

// ---------------------------------------------------------------------------
// Kernel 2: per-patch class histogram -> A matrix (2048 x 256, f16, row-major).
// A[patch][c] = count_c / 256  (exact in f16). One 256-thread WG per patch.
// ---------------------------------------------------------------------------
__global__ __launch_bounds__(256) void hist_kernel(
    const int* __restrict__ smap, _Float16* __restrict__ A) {
  __shared__ unsigned hist[NUM_CLASSES];
  int t = threadIdx.x;
  hist[t] = 0u;
  __syncthreads();
  int patch = blockIdx.x;               // 0..2047
  int b  = patch >> 10;
  int pi = patch & 1023;
  int ph = pi >> 5, pw = pi & 31;
  int y = ph * 16 + (t >> 4);
  int x = pw * 16 + (t & 15);
  int cls = smap[(size_t)b * 512 * 512 + y * 512 + x];
  cls = cls < 0 ? 0 : (cls > 255 ? 255 : cls);
  atomicAdd(&hist[cls], 1u);
  __syncthreads();
  A[(size_t)patch * K_DIM + t] = (_Float16)((float)hist[t] * (1.0f / 256.0f));
}

// ---------------------------------------------------------------------------
// Kernel 3 (fused): WMMA GEMM + sincos pos-embed + LayerNorm -> d_out.
// One block (8 waves) owns a 16-token x 1024-channel strip; wave w owns
// columns [w*128, w*128+128). 64 v_wmma per wave (8 kt x 8 j). LN statistics:
// per-lane per-row partials -> shfl_xor over 16-lane half-groups -> LDS
// cross-wave reduce -> normalize in-register -> store. No pooled round-trip.
// ---------------------------------------------------------------------------
union AFrag { v16h v; v8h h[2]; };

__global__ __launch_bounds__(256) void gemm_posln_kernel(
    const _Float16* __restrict__ A, const _Float16* __restrict__ Bswz,
    const float* __restrict__ gamma, const float* __restrict__ beta,
    float* __restrict__ out) {
  __shared__ float redS[16][8];
  __shared__ float redQ[16][8];
  __shared__ float meanArr[16];
  __shared__ float rstdArr[16];

  int lane = threadIdx.x & 31;
  int wave = threadIdx.x >> 5;           // 0..7 -> 128-column strip
  int mt   = blockIdx.x;                 // 0..127 -> 16-token strip
  int nbase = wave * 128;
  int row  = mt * 16 + (lane & 15);
  int aOff = row * K_DIM + ((lane & 16) ? 8 : 0);
  int mAdd = (lane & 16) ? 8 : 0;        // rows held: m = v + mAdd

  v8f c[8] = {};
#pragma unroll
  for (int kt = 0; kt < 8; ++kt) {
    AFrag a;
    a.h[0] = *(const v8h*)(A + aOff + kt * 32);
    a.h[1] = *(const v8h*)(A + aOff + kt * 32 + 16);
    const _Float16* bb = Bswz + (size_t)(kt * 64 + wave * 8) * 512 + lane * 16;
#pragma unroll
    for (int j = 0; j < 8; ++j) {
      v16h b = *(const v16h*)(bb + j * 512);
      c[j] = __builtin_amdgcn_wmma_f32_16x16x32_f16(false, a.v, false, b,
                                                    (short)0, c[j], false, false);
    }
  }

  // --- pos-embed + per-row partial sums (rows m = v + mAdd) ---
  // token r = mt*16 + m; h = (r & 1023) >> 5 is constant per block;
  // w = (mt&1)*16 + m.
  int hh    = (mt & 63) >> 1;
  int wBase = (mt & 1) * 16;
  float sv[8] = {}, qv[8] = {};
  float g[8], bt[8];
#pragma unroll
  for (int j = 0; j < 8; ++j) {
    int n = nbase + j * 16 + (lane & 15);
    g[j]  = gamma[n];
    bt[j] = beta[n];
    int sect = n >> 8;                    // 0:sin_h 1:cos_h 2:sin_w 3:cos_w
    int jj = n & 255;
    float omega = __expf((float)jj * (-9.210340371976184f / 256.0f));
    if (sect < 2) {
      float ang = (float)hh * omega;
      float pos = (sect & 1) ? __cosf(ang) : __sinf(ang);
#pragma unroll
      for (int v = 0; v < 8; ++v) {
        float val = c[j][v] + pos;
        c[j][v] = val; sv[v] += val; qv[v] += val * val;
      }
    } else {
#pragma unroll
      for (int v = 0; v < 8; ++v) {
        float ang = (float)(wBase + v + mAdd) * omega;
        float pos = (sect & 1) ? __cosf(ang) : __sinf(ang);
        float val = c[j][v] + pos;
        c[j][v] = val; sv[v] += val; qv[v] += val * val;
      }
    }
  }

  // --- cross-lane reduce within each 16-lane half-group (same rows) ---
#pragma unroll
  for (int v = 0; v < 8; ++v) {
#pragma unroll
    for (int msk = 1; msk < 16; msk <<= 1) {
      sv[v] += __shfl_xor(sv[v], msk, 32);
      qv[v] += __shfl_xor(qv[v], msk, 32);
    }
  }
  if ((lane & 15) == 0) {                 // lanes 0 and 16 publish partials
#pragma unroll
    for (int v = 0; v < 8; ++v) {
      redS[v + mAdd][wave] = sv[v];
      redQ[v + mAdd][wave] = qv[v];
    }
  }
  __syncthreads();
  if (threadIdx.x < 16) {
    float ts = 0.f, tq = 0.f;
#pragma unroll
    for (int wv = 0; wv < 8; ++wv) { ts += redS[threadIdx.x][wv]; tq += redQ[threadIdx.x][wv]; }
    float mean = ts * (1.0f / 1024.0f);
    float var  = tq * (1.0f / 1024.0f) - mean * mean;
    meanArr[threadIdx.x] = mean;
    rstdArr[threadIdx.x] = rsqrtf(var + 1e-5f);
  }
  __syncthreads();

  // --- normalize + affine + store ---
#pragma unroll
  for (int v = 0; v < 8; ++v) {
    int m = v + mAdd;
    float mean = meanArr[m];
    float rstd = rstdArr[m];
    float* p = out + (size_t)(mt * 16 + m) * N_DIM + nbase + (lane & 15);
#pragma unroll
    for (int j = 0; j < 8; ++j) {
      p[j * 16] = (c[j][v] - mean) * rstd * g[j] + bt[j];
    }
  }
}

// ---------------------------------------------------------------------------
extern "C" void kernel_launch(void* const* d_in, const int* in_sizes, int n_in,
                              void* d_out, int out_size, void* d_ws, size_t ws_size,
                              hipStream_t stream) {
  const int*   smap  = (const int*)d_in[0];     // (2,1,512,512) int32
  const float* embed = (const float*)d_in[1];   // (256,1024) f32
  const float* gamma = (const float*)d_in[2];   // (1024,) f32
  const float* beta  = (const float*)d_in[3];   // (1024,) f32
  float* out = (float*)d_out;                   // (2,1024,1024) f32

  char* ws = (char*)d_ws;
  _Float16* A    = (_Float16*)ws;               // 2048*256*2 = 1 MB
  _Float16* Bswz = (_Float16*)(ws + (1 << 20)); // 256*1024*2 = 512 KB

  swizzle_b_kernel<<<64,   256, 0, stream>>>(embed, Bswz);
  hist_kernel     <<<2048, 256, 0, stream>>>(smap, A);
  gemm_posln_kernel<<<128, 256, 0, stream>>>(A, Bswz, gamma, beta, out);
}
